// Model_23158463660758
// MI455X (gfx1250) — matrix operand order repaired and checked by reading, not compiled, not run on hardware
//
#include <hip/hip_runtime.h>
#include <hip/hip_bf16.h>
#include <math.h>

// ---------------------------------------------------------------------------
// Types for CDNA5 WMMA
// ---------------------------------------------------------------------------
typedef __attribute__((ext_vector_type(16))) __bf16 v16bf;
typedef __attribute__((ext_vector_type(2)))  __bf16 v2bf;
typedef __attribute__((ext_vector_type(8)))  float  v8f;

// Native f32 -> bf16 conversion (lowers to v_cvt_pk_bf16_f32 on gfx1250).
__device__ __forceinline__ __bf16 f2bf(float f) { return (__bf16)f; }
__device__ __forceinline__ unsigned pk2(float x, float y) {
  v2bf v; v.x = (__bf16)x; v.y = (__bf16)y;
  return __builtin_bit_cast(unsigned, v);
}

// ---------------------------------------------------------------------------
// Tiled GEMM: C[M,N] = A[M,K] @ W[K,N] + bias, optional ReLU.
// BM=128, BN=64, BK=32. 256 threads = 8 wave32; wave w computes rows
// [w*16, w*16+16) x 64 cols via 4 v_wmma_f32_16x16x32_bf16 accumulators.
//
// LDS layouts are fragment-order friendly:
//   As [row m][k]      row stride 40 bf16 (80B)  -> A frag = 2x ds_load_b128
//   BsT[col n][k]      row stride 40 bf16 (80B)  -> B frag = 2x ds_load_b128
// Row starts 20*r mod 64 banks enumerate all multiples of 4 -> b128 reads of a
// 16-lane group cover all 64 banks exactly once (conflict-free).
//
// Software pipeline, ping-pong LDS (one barrier per k-step):
//   issue global loads for tile s+1 -> load ALL fragments (10x ds_load_b128,
//   distinct regs so waits can be partial) -> 4 WMMAs -> wait loads, store
//   tile s+1 into buffer (s+1)&1 -> barrier.
// ---------------------------------------------------------------------------
#define BM 128
#define BN 64
#define BK 32
#define LDP 40   // padded LDS row stride in bf16 elements

__global__ __launch_bounds__(256) void gemm_bf16_kernel(
    const float* __restrict__ A, const float* __restrict__ W,
    const float* __restrict__ bias, float* __restrict__ C,
    int M, int N, int K, int relu)
{
  __shared__ __align__(16) __bf16 As[2][BM][LDP];    // 2*10240 B
  __shared__ __align__(16) __bf16 BsT[2][BN][LDP];   // 2* 5120 B

  const int tid  = threadIdx.x;
  const int wave = tid >> 5;
  const int lane = tid & 31;
  const int m0   = blockIdx.y * BM;
  const int n0   = blockIdx.x * BN;

  v8f acc[4];
#pragma unroll
  for (int nt = 0; nt < 4; ++nt)
#pragma unroll
    for (int r = 0; r < 8; ++r) acc[nt][r] = 0.0f;

  // cooperative-load mapping
  const int ar = tid >> 1;           // 0..127  (A row)
  const int ac = (tid & 1) * 16;     // 0 or 16 (A col base)
  const int br = tid >> 3;           // 0..31   (W k-row)
  const int bc = (tid & 7) * 8;      // 0..56   (W n-col base)

  const int l15  = lane & 15;
  const int mrow = wave * 16 + l15;
  const int hl   = lane >> 4;        // half: 0 or 1

  const float* apBase = A + (size_t)(m0 + ar) * K + ac;
  const float* bpBase = W + (size_t)br * N + n0 + bc;
  const size_t bStep  = (size_t)BK * N;
  const int    ksteps = K / BK;

  struct B32x2 { uint4 a, b; };      // 32B container for a v16bf fragment

  float4 fa0, fa1, fa2, fa3, fb0, fb1;
  // prologue: stage tile 0 into registers, then into LDS buffer 0
  {
    const float* ap = apBase;
    fa0 = *(const float4*)(ap + 0);
    fa1 = *(const float4*)(ap + 4);
    fa2 = *(const float4*)(ap + 8);
    fa3 = *(const float4*)(ap + 12);
    const float* bp = bpBase;
    fb0 = *(const float4*)(bp + 0);
    fb1 = *(const float4*)(bp + 4);

    *(uint2*)&As[0][ar][ac + 0]  = make_uint2(pk2(fa0.x, fa0.y), pk2(fa0.z, fa0.w));
    *(uint2*)&As[0][ar][ac + 4]  = make_uint2(pk2(fa1.x, fa1.y), pk2(fa1.z, fa1.w));
    *(uint2*)&As[0][ar][ac + 8]  = make_uint2(pk2(fa2.x, fa2.y), pk2(fa2.z, fa2.w));
    *(uint2*)&As[0][ar][ac + 12] = make_uint2(pk2(fa3.x, fa3.y), pk2(fa3.z, fa3.w));
    BsT[0][bc + 0][br] = f2bf(fb0.x);
    BsT[0][bc + 1][br] = f2bf(fb0.y);
    BsT[0][bc + 2][br] = f2bf(fb0.z);
    BsT[0][bc + 3][br] = f2bf(fb0.w);
    BsT[0][bc + 4][br] = f2bf(fb1.x);
    BsT[0][bc + 5][br] = f2bf(fb1.y);
    BsT[0][bc + 6][br] = f2bf(fb1.z);
    BsT[0][bc + 7][br] = f2bf(fb1.w);
  }
  __syncthreads();

  for (int s = 0; s < ksteps; ++s) {
    const int cur = s & 1;
    const int nxt = cur ^ 1;

    // ---- issue global loads for next tile (overlap with WMMAs below) ----
    const bool more = (s + 1 < ksteps);
    if (more) {
      const float* ap = apBase + (size_t)(s + 1) * BK;
      fa0 = *(const float4*)(ap + 0);
      fa1 = *(const float4*)(ap + 4);
      fa2 = *(const float4*)(ap + 8);
      fa3 = *(const float4*)(ap + 12);
      const float* bp = bpBase + (size_t)(s + 1) * bStep;
      fb0 = *(const float4*)(bp + 0);
      fb1 = *(const float4*)(bp + 4);
    }

    // ---- load ALL fragments first (distinct regs -> partial dscnt waits) --
    B32x2 ua;
    ua.a = *(const uint4*)&As[cur][mrow][hl * 8];
    ua.b = *(const uint4*)&As[cur][mrow][16 + hl * 8];
    B32x2 ub[4];
#pragma unroll
    for (int nt = 0; nt < 4; ++nt) {
      const int n = nt * 16 + l15;
      ub[nt].a = *(const uint4*)&BsT[cur][n][hl * 8];
      ub[nt].b = *(const uint4*)&BsT[cur][n][16 + hl * 8];
    }
    const v16bf afr = __builtin_bit_cast(v16bf, ua);

    // ---- 4 WMMAs ----
#pragma unroll
    for (int nt = 0; nt < 4; ++nt) {
      const v16bf bfr = __builtin_bit_cast(v16bf, ub[nt]);
      acc[nt] = __builtin_amdgcn_wmma_f32_16x16x32_bf16(
          false, afr, false, bfr, (short)0, acc[nt], false, false);
    }

    // ---- stage next tile into the other LDS buffer ----
    if (more) {
      *(uint2*)&As[nxt][ar][ac + 0]  = make_uint2(pk2(fa0.x, fa0.y), pk2(fa0.z, fa0.w));
      *(uint2*)&As[nxt][ar][ac + 4]  = make_uint2(pk2(fa1.x, fa1.y), pk2(fa1.z, fa1.w));
      *(uint2*)&As[nxt][ar][ac + 8]  = make_uint2(pk2(fa2.x, fa2.y), pk2(fa2.z, fa2.w));
      *(uint2*)&As[nxt][ar][ac + 12] = make_uint2(pk2(fa3.x, fa3.y), pk2(fa3.z, fa3.w));
      BsT[nxt][bc + 0][br] = f2bf(fb0.x);
      BsT[nxt][bc + 1][br] = f2bf(fb0.y);
      BsT[nxt][bc + 2][br] = f2bf(fb0.z);
      BsT[nxt][bc + 3][br] = f2bf(fb0.w);
      BsT[nxt][bc + 4][br] = f2bf(fb1.x);
      BsT[nxt][bc + 5][br] = f2bf(fb1.y);
      BsT[nxt][bc + 6][br] = f2bf(fb1.z);
      BsT[nxt][bc + 7][br] = f2bf(fb1.w);
    }
    __syncthreads();
  }

  // epilogue: C/D layout — VGPR r: lanes 0-15 -> M=r, lanes 16-31 -> M=r+8
#pragma unroll
  for (int nt = 0; nt < 4; ++nt) {
    const int ncol = n0 + nt * 16 + l15;
    const float bv = bias ? bias[ncol] : 0.0f;
#pragma unroll
    for (int r = 0; r < 8; ++r) {
      const int mr = m0 + wave * 16 + hl * 8 + r;
      float v = acc[nt][r] + bv;
      if (relu) v = fmaxf(v, 0.0f);
      C[(size_t)mr * N + ncol] = v;
    }
  }
}

// ---------------------------------------------------------------------------
// Embedding + positional encoding: out[r,d] = emb[tok[r],d]*sqrt(D) + pe(pos,d)
// pe[pos,2j] = sin(pos*10000^(-4j/512)), pe[pos,2j+1] = cos(...)
// ---------------------------------------------------------------------------
__global__ __launch_bounds__(256) void embed_kernel(
    const float* __restrict__ emb, const int* __restrict__ tok,
    float* __restrict__ out)
{
  const int r   = blockIdx.x;          // token 0..2047
  const int pos = r & 1023;
  const int t   = threadIdx.x;
  const int id  = tok[r];
  const float scale = 22.627416997969522f;  // sqrt(512)
#pragma unroll
  for (int i = 0; i < 2; ++i) {
    const int d = t + 256 * i;
    const int j = d >> 1;
    const float den = powf(10000.0f, -4.0f * (float)j / 512.0f);
    const float ang = (float)pos * den;
    const float pe  = (d & 1) ? cosf(ang) : sinf(ang);
    out[(size_t)r * 512 + d] = emb[(size_t)id * 512 + d] * scale + pe;
  }
}

// ---------------------------------------------------------------------------
// Fused residual add + LayerNorm over D=512. One wave32 per row, 8 rows/block.
// ---------------------------------------------------------------------------
__global__ __launch_bounds__(256) void ln_add_kernel(
    const float* __restrict__ a, const float* __restrict__ b,
    const float* __restrict__ g, const float* __restrict__ beta,
    float* __restrict__ out)
{
  const int wave = threadIdx.x >> 5;
  const int lane = threadIdx.x & 31;
  const int row  = blockIdx.x * 8 + wave;
  const float* pa = a + (size_t)row * 512;
  const float* pb = b + (size_t)row * 512;

  float x[16];
  float s = 0.0f;
#pragma unroll
  for (int i = 0; i < 16; ++i) {
    x[i] = pa[lane + 32 * i] + pb[lane + 32 * i];
    s += x[i];
  }
#pragma unroll
  for (int off = 16; off > 0; off >>= 1) s += __shfl_xor(s, off, 32);
  const float mu = s * (1.0f / 512.0f);

  float v = 0.0f;
#pragma unroll
  for (int i = 0; i < 16; ++i) { const float d = x[i] - mu; v += d * d; }
#pragma unroll
  for (int off = 16; off > 0; off >>= 1) v += __shfl_xor(v, off, 32);
  const float rstd = rsqrtf(v * (1.0f / 512.0f) + 1e-5f);

  float* po = out + (size_t)row * 512;
#pragma unroll
  for (int i = 0; i < 16; ++i) {
    const int d = lane + 32 * i;
    po[d] = (x[i] - mu) * rstd * g[d] + beta[d];
  }
}

// ---------------------------------------------------------------------------
// Non-causal linear attention, stage 1: per (b,h) compute
//   kn = softmax over sequence (axis n) of k;  ctx[d][e] = sum_n kn[n,d]*v[n,e]
// One 64-thread block per (b,h); thread e keeps ctx[:,e] in registers.
// ---------------------------------------------------------------------------
__global__ __launch_bounds__(64) void linctx_kernel(
    const float* __restrict__ base, int stride, int koff, int voff,
    float* __restrict__ ctx)
{
  const int bh = blockIdx.x;
  const int b  = bh >> 3, h = bh & 7;
  const int t  = threadIdx.x;   // doubles as column d (pass 1) and e (pass 2)

  const float* kcol = base + (size_t)b * 1024 * stride + koff + h * 64 + t;
  float mx = -1e30f;
  for (int i = 0; i < 1024; ++i) mx = fmaxf(mx, kcol[(size_t)i * stride]);
  float sm = 0.0f;
  for (int i = 0; i < 1024; ++i) sm += expf(kcol[(size_t)i * stride] - mx);
  const float inv = 1.0f / sm;

  __shared__ float kn[64];
  float creg[64];
#pragma unroll
  for (int d = 0; d < 64; ++d) creg[d] = 0.0f;

  const float* vrow0 = base + (size_t)b * 1024 * stride + voff + h * 64;
  for (int i = 0; i < 1024; ++i) {
    __syncthreads();
    kn[t] = expf(kcol[(size_t)i * stride] - mx) * inv;
    __syncthreads();
    const float ve = vrow0[(size_t)i * stride + t];
#pragma unroll
    for (int d = 0; d < 64; ++d) creg[d] += kn[d] * ve;
  }
  float* cp = ctx + (size_t)bh * 4096;
#pragma unroll
  for (int d = 0; d < 64; ++d) cp[d * 64 + t] = creg[d];
}

// ---------------------------------------------------------------------------
// Non-causal linear attention, stage 2: per (token, head):
//   qs = softmax(q_row)*dh^-0.5 ; out[e] = sum_d qs[d]*ctx[d][e]
// ---------------------------------------------------------------------------
__global__ __launch_bounds__(64) void linout_kernel(
    const float* __restrict__ base, int stride, int qoff,
    const float* __restrict__ ctx, float* __restrict__ out)
{
  const int h = blockIdx.x;       // 0..7
  const int r = blockIdx.y;       // 0..2047
  const int b = r >> 10;
  const int t = threadIdx.x;      // 0..63

  __shared__ float red[64];
  __shared__ float qs[64];

  const float q = base[(size_t)r * stride + qoff + h * 64 + t];
  red[t] = q; __syncthreads();
  for (int s = 32; s > 0; s >>= 1) {
    if (t < s) red[t] = fmaxf(red[t], red[t + s]);
    __syncthreads();
  }
  const float m = red[0]; __syncthreads();
  const float e = expf(q - m);
  red[t] = e; __syncthreads();
  for (int s = 32; s > 0; s >>= 1) {
    if (t < s) red[t] += red[t + s];
    __syncthreads();
  }
  qs[t] = e * (0.125f / red[0]);   // dh^-0.5 = 1/8
  __syncthreads();

  const float* cp = ctx + (size_t)(b * 8 + h) * 4096;
  float o = 0.0f;
#pragma unroll
  for (int d = 0; d < 64; ++d) o += qs[d] * cp[d * 64 + t];
  out[(size_t)r * 512 + h * 64 + t] = o;
}

// ---------------------------------------------------------------------------
// Causal linear attention: per (b,h) sequential scan. Thread e keeps the
// 64x64 running k-outer-v state column S[:,e] in registers; kc per d=t.
// qvk row layout: q at +0, v at +512, k at +1024, stride 1536.
// ---------------------------------------------------------------------------
__global__ __launch_bounds__(64) void causal_kernel(
    const float* __restrict__ qvk, float* __restrict__ out)
{
  const int bh = blockIdx.x;
  const int b  = bh >> 3, h = bh & 7;
  const int t  = threadIdx.x;

  __shared__ float red[64];
  __shared__ float qs[64];
  __shared__ float ke[64];

  float S[64];
#pragma unroll
  for (int d = 0; d < 64; ++d) S[d] = 0.0f;
  float kc = 0.0f;

  const float* rowbase = qvk + (size_t)b * 1024 * 1536 + h * 64;
  for (int n = 0; n < 1024; ++n) {
    const float* row = rowbase + (size_t)n * 1536;
    const float q  = row[t];
    const float kv = row[1024 + t];
    const float vv = row[512 + t];

    // row softmax of q
    red[t] = q; __syncthreads();
    for (int s = 32; s > 0; s >>= 1) {
      if (t < s) red[t] = fmaxf(red[t], red[t + s]);
      __syncthreads();
    }
    const float m = red[0]; __syncthreads();
    const float e = expf(q - m);
    red[t] = e; __syncthreads();
    for (int s = 32; s > 0; s >>= 1) {
      if (t < s) red[t] += red[t + s];
      __syncthreads();
    }
    const float qsv = e * (0.125f / red[0]);
    __syncthreads();

    const float kev = expf(kv);
    kc += kev;
    qs[t]  = qsv;
    ke[t]  = kev;
    red[t] = (kc + 1e-6f) * qsv;    // denom contribution for d=t
    __syncthreads();
    for (int s = 32; s > 0; s >>= 1) {
      if (t < s) red[t] += red[t + s];
      __syncthreads();
    }
    const float dinv = 1.0f / red[0];

    float num = 0.0f;
#pragma unroll
    for (int d = 0; d < 64; ++d) {
      S[d] += ke[d] * vv;           // inclusive cumsum of k (x) v
      num  += qs[d] * S[d];
    }
    out[((size_t)b * 1024 + n) * 512 + h * 64 + t] = num * dinv;
    __syncthreads();
  }
}

// ---------------------------------------------------------------------------
// Orchestration.
// Input flattening (setup_inputs dict insertion order):
//   0 emb_x, 1 emb_y,
//   enc l (base=2+10l):  qvk.w,qvk.b, ln1.g,ln1.b, ff1.w,ff1.b, ff2.w,ff2.b, ln2.g,ln2.b
//   dec l (base=62+16l): qvk.w,qvk.b, ln1.g,ln1.b, kv.w,kv.b, q.w,q.b,
//                        ln2.g,ln2.b, ff1.w,ff1.b, ff2.w,ff2.b, ln3.g,ln3.b
//   158 gen1.w, 159 gen1.b, 160 gen2.w, 161 gen2.b, 162 x, 163 y
// ---------------------------------------------------------------------------
#define IN(i) ((const float*)d_in[(i)])

extern "C" void kernel_launch(void* const* d_in, const int* in_sizes, int n_in,
                              void* d_out, int out_size, void* d_ws, size_t ws_size,
                              hipStream_t stream) {
  (void)in_sizes; (void)n_in; (void)out_size; (void)ws_size;

  const int* x = (const int*)d_in[162];
  const int* y = (const int*)d_in[163];

  float* actZ    = (float*)d_ws;                 // 2048*512
  float* actH    = actZ    + 2048 * 512;         // 2048*512
  float* bufQVK  = actH    + 2048 * 512;         // 2048*1536
  float* bufFF   = bufQVK  + 2048 * 1536;        // 2048*2048
  float* bufAttn = bufFF   + 2048 * 2048;        // 2048*512
  float* bufTmp  = bufAttn + 2048 * 512;         // 2048*512
  float* bufKV   = bufTmp  + 2048 * 512;         // 2048*1024
  float* bufCtx  = bufKV   + 2048 * 1024;        // 16*64*64

  auto gemm = [&](const float* A, const float* W, const float* bias, float* C,
                  int M, int N, int K, int relu) {
    dim3 g(N / BN, M / BM);
    gemm_bf16_kernel<<<g, 256, 0, stream>>>(A, W, bias, C, M, N, K, relu);
  };
  auto lnadd = [&](const float* a, const float* b, const float* g,
                   const float* be, float* o) {
    ln_add_kernel<<<256, 256, 0, stream>>>(a, b, g, be, o);
  };

  // embeddings + positional encoding
  embed_kernel<<<2048, 256, 0, stream>>>(IN(0), x, actZ);
  embed_kernel<<<2048, 256, 0, stream>>>(IN(1), y, actH);

  // ---------------- encoder ----------------
  for (int l = 0; l < 6; ++l) {
    const int p = 2 + 10 * l;
    gemm(actZ, IN(p + 0), IN(p + 1), bufQVK, 2048, 1536, 512, 0);
    linctx_kernel<<<16, 64, 0, stream>>>(bufQVK, 1536, 1024, 512, bufCtx);
    linout_kernel<<<dim3(8, 2048), 64, 0, stream>>>(bufQVK, 1536, 0, bufCtx, bufAttn);
    lnadd(bufAttn, actZ, IN(p + 2), IN(p + 3), actZ);
    gemm(actZ, IN(p + 4), IN(p + 5), bufFF, 2048, 2048, 512, 1);
    gemm(bufFF, IN(p + 6), IN(p + 7), bufTmp, 2048, 512, 2048, 0);
    lnadd(bufTmp, actZ, IN(p + 8), IN(p + 9), actZ);
  }

  // ---------------- decoder ----------------
  for (int l = 0; l < 6; ++l) {
    const int p = 62 + 16 * l;
    gemm(actH, IN(p + 0), IN(p + 1), bufQVK, 2048, 1536, 512, 0);
    causal_kernel<<<16, 64, 0, stream>>>(bufQVK, bufAttn);
    lnadd(bufAttn, actH, IN(p + 2), IN(p + 3), actH);
    // cross attention: kv from encoder memory, q2 from current state
    gemm(actZ, IN(p + 4), IN(p + 5), bufKV, 2048, 1024, 512, 0);
    gemm(actH, IN(p + 6), IN(p + 7), bufTmp, 2048, 512, 512, 0);
    linctx_kernel<<<16, 64, 0, stream>>>(bufKV, 1024, 0, 512, bufCtx);
    linout_kernel<<<dim3(8, 2048), 64, 0, stream>>>(bufTmp, 512, 0, bufCtx, bufAttn);
    lnadd(bufAttn, actH, IN(p + 8), IN(p + 9), actH);
    gemm(actH, IN(p + 10), IN(p + 11), bufFF, 2048, 2048, 512, 1);
    gemm(bufFF, IN(p + 12), IN(p + 13), bufTmp, 2048, 512, 2048, 0);
    lnadd(bufTmp, actH, IN(p + 14), IN(p + 15), actH);
  }

  // ---------------- generator ----------------
  gemm(actH, IN(158), IN(159), bufFF, 2048, 2048, 512, 1);
  gemm(bufFF, IN(160), IN(161), (float*)d_out, 2048, 32000, 2048, 0);
}
